// NeuralHMMClassifier_17008070492631
// MI455X (gfx1250) — matrix operand order repaired and checked by reading, not compiled, not run on hardware
//
#include <hip/hip_runtime.h>
#include <hip/hip_bf16.h>
#include <math.h>

typedef __attribute__((ext_vector_type(2)))  float  v2f;
typedef __attribute__((ext_vector_type(8)))  float  v8f;
typedef __attribute__((ext_vector_type(16))) __bf16 v16bf;

#define KK 128
#define VV 50257
#define DD 128
#define HH 128
#define BB 64
#define TT 2048
#define NVT ((VV + 15) / 16)      // 3142 vocab tiles of 16 rows
#define SROWS 50272               // vocab rounded up to 16
#define NZB 50                    // logZ partial blocks (1024 rows each)

// ---------------- zero output ----------------
__global__ void k_zero(float* out) { if (threadIdx.x == 0) out[0] = 0.f; }

// ---------------- transition matrix: A[i][j] = softmax_j(W@q + b) ----------------
__global__ void k_transA(const float* __restrict__ W, const float* __restrict__ q,
                         const float* __restrict__ bvec, float* __restrict__ Amat) {
    int i = blockIdx.x, j = threadIdx.x;
    const float* row = W + (size_t)(i * KK + j) * HH;
    float acc = bvec[i * KK + j];
    for (int h = 0; h < HH; ++h) acc += row[h] * q[h];
    __shared__ float sm[KK];
    sm[j] = acc; __syncthreads();
    for (int o = 64; o > 0; o >>= 1) { if (j < o) sm[j] = fmaxf(sm[j], sm[j + o]); __syncthreads(); }
    float m = sm[0]; __syncthreads();
    sm[j] = expf(acc - m); __syncthreads();
    for (int o = 64; o > 0; o >>= 1) { if (j < o) sm[j] += sm[j + o]; __syncthreads(); }
    float lse = m + logf(sm[0]);
    Amat[i * KK + j] = expf(acc - lse);   // probability-space transition
}

// ---------------- initial distribution pi ----------------
__global__ void k_pi(const float* __restrict__ logits, float* __restrict__ pi) {
    int j = threadIdx.x;
    float x = logits[j];
    __shared__ float sm[KK];
    sm[j] = x; __syncthreads();
    for (int o = 64; o > 0; o >>= 1) { if (j < o) sm[j] = fmaxf(sm[j], sm[j + o]); __syncthreads(); }
    float m = sm[0]; __syncthreads();
    sm[j] = expf(x - m); __syncthreads();
    for (int o = 64; o > 0; o >>= 1) { if (j < o) sm[j] += sm[j + o]; __syncthreads(); }
    pi[j] = expf(x - m - logf(sm[0]));
}

// ---------------- emission scores S[v][k] = we[v]·te[k] + bias[v] via bf16 WMMA ----------------
__global__ void __launch_bounds__(256) k_scores(const float* __restrict__ we,
                                                const float* __restrict__ te,
                                                const float* __restrict__ bias,
                                                float* __restrict__ S) {
    int wave = threadIdx.x >> 5, lane = threadIdx.x & 31;
    int vt = blockIdx.x * 8 + wave;
    if (vt >= NVT) return;                 // wave-uniform exit, no barriers in this kernel
    int vbase = vt * 16;
    int h = lane >> 4, n = lane & 15, m = lane & 15;
    int arow = vbase + m; if (arow >= VV) arow = VV - 1;

    // A fragments: 16 vocab rows x K=128 (4 chunks of 32), bf16
    v16bf af[4];
    #pragma unroll
    for (int c = 0; c < 4; ++c) {
        int kc = c * 32;
        #pragma unroll
        for (int e = 0; e < 16; ++e) {
            int kA = kc + (e < 8 ? e : e + 8) + 8 * h;     // ISA 16-bit A layout
            af[c][e] = (__bf16)we[(size_t)arow * DD + kA];
        }
    }
    // loop over 8 tiles of 16 tag columns
    for (int nt = 0; nt < 8; ++nt) {
        int trow = nt * 16 + n;
        v8f acc = {};
        #pragma unroll
        for (int c = 0; c < 4; ++c) {
            int kc = c * 32;
            v16bf bf;
            #pragma unroll
            for (int e = 0; e < 16; ++e)                    // ISA 16-bit B layout: K = (e<8 ? e : e+8) + 8h
                bf[e] = (__bf16)te[(size_t)trow * DD + kc + (e < 8 ? e : e + 8) + 8 * h];
            acc = __builtin_amdgcn_wmma_f32_16x16x32_bf16(false, af[c], false, bf,
                                                          (short)0, acc, false, false);
        }
        int kcol = nt * 16 + n;
        #pragma unroll
        for (int r = 0; r < 8; ++r) {
            int v = vbase + r + 8 * h;                      // C layout: M = r + 8h, N = lane%16
            if (v < VV) S[(size_t)v * KK + kcol] = acc[r] + bias[v];
        }
    }
}

// ---------------- logZ[k] = logsumexp_v S[v][k] : partials then combine ----------------
__global__ void k_logZpart(const float* __restrict__ S, float2* __restrict__ part) {
    int k = threadIdx.x & 127, par = threadIdx.x >> 7;
    int vbase = blockIdx.x * 1024;
    int vend = vbase + 1024; if (vend > VV) vend = VV;
    float m = -__builtin_inff(), s = 0.f;
    for (int v = vbase + par; v < vend; v += 2) {
        float x = S[(size_t)v * KK + k];
        if (x > m) { s = s * expf(m - x) + 1.f; m = x; } else { s += expf(x - m); }
    }
    __shared__ float sm[2][128], ss[2][128];
    sm[par][k] = m; ss[par][k] = s; __syncthreads();
    if (par == 0) {
        float m2 = sm[1][k], s2 = ss[1][k];
        float M = fmaxf(m, m2);
        part[(size_t)blockIdx.x * 128 + k] = make_float2(M, s * expf(m - M) + s2 * expf(m2 - M));
    }
}

__global__ void k_logZ2(const float2* __restrict__ part, float* __restrict__ logZ) {
    int k = threadIdx.x;
    float M = -__builtin_inff(), s = 0.f;
    for (int b = 0; b < NZB; ++b) {
        float2 p = part[(size_t)b * 128 + k];
        float nm = fmaxf(M, p.x);
        s = s * expf(M - nm) + p.y * expf(p.x - nm);
        M = nm;
    }
    logZ[k] = M + logf(s);
}

// ---------------- gather per-token emission probs: eexp[t][b][k] ----------------
__global__ void k_emit(const int* __restrict__ tok, const float* __restrict__ S,
                       const float* __restrict__ logZ, float* __restrict__ eexp) {
    int idx = blockIdx.x;              // b*T + t
    int b = idx / TT, t = idx % TT;
    int w = tok[idx];
    int k = threadIdx.x;
    eexp[((size_t)t * BB + b) * KK + k] = expf(S[(size_t)w * KK + k] - logZ[k]);
}

// ---------------- scaled forward recursion: f32 WMMA 16x16x4, pipelined ----------------
__global__ void __launch_bounds__(256) k_forward(const float* __restrict__ Amat,
                                                 const float* __restrict__ pi,
                                                 const float* __restrict__ eexp,
                                                 float* __restrict__ out) {
    __shared__ float pbuf[2][16][132];   // unnormalized forward vars, padded rows
    __shared__ float red[16][16];
    __shared__ float invS[16];
    __shared__ float cacc[16];
    int tid = threadIdx.x;
    int bbase = blockIdx.x * 16;
    int lane = tid & 31, h = lane >> 4, n = lane & 15, m = lane & 15;
    int jbase = (tid >> 5) * 16;         // wave -> 16-wide j tile

    // preload static B fragments (transition probs), 32 K-chunks of 4
    v2f bfr[32];
    #pragma unroll
    for (int c = 0; c < 32; ++c) {
        int kk = c * 4 + 2 * h;          // f32 16x16x4 B layout: K = e + 2h
        bfr[c].x = Amat[(size_t)kk * KK + jbase + n];
        bfr[c].y = Amat[(size_t)(kk + 1) * KK + jbase + n];
    }
    if (tid < 16) cacc[tid] = 0.f;
    int b = tid >> 4, seg = tid & 15;
    // t = 0: u_0 = pi_j * eexp0 (kept unnormalized; scale deferred one step)
    for (int jj = 0; jj < 8; ++jj) {
        int j = seg * 8 + jj;
        pbuf[0][b][j] = pi[j] * eexp[(size_t)(bbase + b) * KK + j];
    }
    __syncthreads();
    {   // normalizer of u_0
        float part = 0.f;
        #pragma unroll
        for (int jj = 0; jj < 8; ++jj) part += pbuf[0][b][seg * 8 + jj];
        red[b][seg] = part; __syncthreads();
        if (tid < 16) {
            float Sb = 0.f;
            #pragma unroll
            for (int s2 = 0; s2 < 16; ++s2) Sb += red[tid][s2];
            cacc[tid] += logf(Sb);
            invS[tid] = 1.f / Sb;
        }
        __syncthreads();
    }
    // prefetch emissions for t = 1
    float epf[8];
    #pragma unroll
    for (int r = 0; r < 8; ++r) {
        int mb = r + 8 * h;
        epf[r] = eexp[((size_t)1 * BB + bbase + mb) * KK + jbase + n];
    }

    for (int t = 1; t < TT; ++t) {
        int cur = (t - 1) & 1, nxt = t & 1;
        // prefetch t+1 emissions; latency hides under the WMMA chain
        float enext[8] = {};
        if (t + 1 < TT) {
            #pragma unroll
            for (int r = 0; r < 8; ++r) {
                int mb = r + 8 * h;
                enext[r] = eexp[((size_t)(t + 1) * BB + bbase + mb) * KK + jbase + n];
            }
        }
        // q = u_{t-1} x A  with two independent accumulator chains
        v8f acc0 = {}, acc1 = {};
        #pragma unroll
        for (int c = 0; c < 32; c += 2) {
            int kk = c * 4 + 2 * h;      // f32 16x16x4 A layout: M = lane%16, K = e + 2h
            v2f af0, af1;
            af0.x = pbuf[cur][m][kk];     af0.y = pbuf[cur][m][kk + 1];
            af1.x = pbuf[cur][m][kk + 4]; af1.y = pbuf[cur][m][kk + 5];
            acc0 = __builtin_amdgcn_wmma_f32_16x16x4_f32(false, af0, false, bfr[c],
                                                         (short)0, acc0, false, false);
            acc1 = __builtin_amdgcn_wmma_f32_16x16x4_f32(false, af1, false, bfr[c + 1],
                                                         (short)0, acc1, false, false);
        }
        // u_t = q * invS_{t-1} * e_t   (C layout: M = r + 8h, N = lane%16)
        #pragma unroll
        for (int r = 0; r < 8; ++r) {
            int mb = r + 8 * h;
            pbuf[nxt][mb][jbase + n] = (acc0[r] + acc1[r]) * epf[r] * invS[mb];
        }
        __syncthreads();
        // normalizer of u_t
        float part = 0.f;
        #pragma unroll
        for (int jj = 0; jj < 8; ++jj) part += pbuf[nxt][b][seg * 8 + jj];
        red[b][seg] = part; __syncthreads();
        if (tid < 16) {
            float Sb = 0.f;
            #pragma unroll
            for (int s2 = 0; s2 < 16; ++s2) Sb += red[tid][s2];
            cacc[tid] += logf(Sb);
            invS[tid] = 1.f / Sb;
        }
        __syncthreads();
        if (t + 1 < TT) {
            #pragma unroll
            for (int r = 0; r < 8; ++r) epf[r] = enext[r];
        }
    }
    if (tid < 16) atomicAdd(out, -cacc[tid]);   // ll_b = sum_t log S_t
}

// ---------------- host launch ----------------
extern "C" void kernel_launch(void* const* d_in, const int* in_sizes, int n_in,
                              void* d_out, int out_size, void* d_ws, size_t ws_size,
                              hipStream_t stream) {
    (void)in_sizes; (void)n_in; (void)out_size; (void)ws_size;
    const int*   tok  = (const int*)d_in[0];
    const float* il   = (const float*)d_in[1];
    const float* tag  = (const float*)d_in[2];
    const float* we   = (const float*)d_in[3];
    const float* bias = (const float*)d_in[4];
    const float* q    = (const float*)d_in[5];
    const float* W    = (const float*)d_in[6];
    const float* tb   = (const float*)d_in[7];
    float* out = (float*)d_out;

    float*  ws   = (float*)d_ws;
    float*  Amat = ws;                                    // 16384 f32
    float*  pi   = Amat + 16384;                          // 128
    float*  logZ = pi + 128;                              // 128
    float2* part = (float2*)(logZ + 128);                 // 50*128 float2
    float*  S    = ws + 32768;                            // SROWS*128 f32 (~25.7 MB)
    float*  eexp = S + (size_t)SROWS * KK;                // T*B*K f32 (~67 MB)

    k_zero    <<<1, 32, 0, stream>>>(out);
    k_transA  <<<128, 128, 0, stream>>>(W, q, tb, Amat);
    k_pi      <<<1, 128, 0, stream>>>(il, pi);
    k_scores  <<<(NVT + 7) / 8, 256, 0, stream>>>(we, tag, bias, S);
    k_logZpart<<<NZB, 256, 0, stream>>>(S, part);
    k_logZ2   <<<1, 128, 0, stream>>>(part, logZ);
    k_emit    <<<BB * TT, 128, 0, stream>>>(tok, S, logZ, eexp);
    k_forward <<<4, 256, 0, stream>>>(Amat, pi, eexp, out);
}
